// OneBodyMLPFieldReadout_55551107007092
// MI455X (gfx1250) — matrix operand order repaired and checked by reading, not compiled, not run on hardware
//
#include <hip/hip_runtime.h>
#include <math.h>

typedef __attribute__((ext_vector_type(16))) _Float16 v16h;
typedef __attribute__((ext_vector_type(8)))  float    v8f;

#define C_DIM 128
#define PATH_W_C (1.0f/16.0f)
#define INV_SQRT3 0.57735026918962576f

// workspace layout (bytes):
//   [0,512)     cq      [512,1024)  cqb
//   [1024,1536) cv      [1536,2048) cvb
//   [2048,2560) cw
//   [2560, ...) f16 weight fragments (units: halfs from frag base):
#define FRAG_OFF_BYTES 2560
#define FR1_OFF 0        // W1: 8 kb * 8 nt * 32 lanes * 16 = 32768 halfs
#define FR2_OFF 32768    // W2: 4*8*512 = 16384
#define FR3_OFF 49152    // W3: 16384
#define FR4_OFF 65536    // W4: 4 kb * 1 nt * 512 = 2048 (cols 1..15 zero-padded)
#define FRAG_TOTAL 67584

// ---------------- prep: rank-1 weight folds ----------------
__global__ void prep_consts(const float* __restrict__ wq_s, const float* __restrict__ Wq0,
                            const float* __restrict__ bq,
                            const float* __restrict__ wv_s, const float* __restrict__ Wv0,
                            const float* __restrict__ bv,
                            const float* __restrict__ wv_v, const float* __restrict__ Wv1,
                            float* __restrict__ wsf) {
  int u = threadIdx.x;
  float cq = 0.f, cqb = 0.f, cv = 0.f, cvb = 0.f, cw = 0.f;
  const float* rq  = wq_s + u * C_DIM;
  const float* rvs = wv_s + u * C_DIM;
  const float* rvv = wv_v + u * C_DIM;
  for (int v = 0; v < C_DIM; ++v) {
    cq  += rq[v]  * Wq0[v];
    cqb += rq[v]  * bq[v];
    cv  += rvs[v] * Wv0[v];
    cvb += rvs[v] * bv[v];
    cw  += rvv[v] * Wv1[v];
  }
  wsf[u] = cq; wsf[128 + u] = cqb; wsf[256 + u] = cv; wsf[384 + u] = cvb; wsf[512 + u] = cw;
}

// ---------------- prep: f16 B-fragment repack ----------------
// B layout (wave32, 16-bit, 32x16 K-block): lane -> column n = nt*16 + (lane&15),
// element e -> k = kb*32 + (e<8 ? (lane>>4)*8 + e : 16 + (lane>>4)*8 + (e-8)). Each lane's 16 halfs are contiguous.
__global__ void prep_frags(const float* __restrict__ W1, const float* __restrict__ W2,
                           const float* __restrict__ W3, const float* __restrict__ W4,
                           unsigned short* __restrict__ fragsRaw) {
  _Float16* frags = reinterpret_cast<_Float16*>(fragsRaw);
  int idx = blockIdx.x * blockDim.x + threadIdx.x;
  if (idx >= FRAG_TOTAL) return;
  const float* W; int Nout; int base;
  if (idx < FR2_OFF)      { W = W1; Nout = 128; base = idx - FR1_OFF; }
  else if (idx < FR3_OFF) { W = W2; Nout = 128; base = idx - FR2_OFF; }
  else if (idx < FR4_OFF) { W = W3; Nout = 128; base = idx - FR3_OFF; }
  else                    { W = W4; Nout = 1;   base = idx - FR4_OFF; }
  int ntiles = (Nout == 1) ? 1 : 8;
  int perKb  = ntiles * 512;
  int kb   = base / perKb;
  int r2   = base % perKb;
  int nt   = r2 / 512;
  int r3   = r2 % 512;
  int lane = r3 / 16;
  int e    = r3 % 16;
  int hi   = lane >> 4;
  int k    = kb * 32 + ((e < 8) ? (hi * 8 + e) : (16 + hi * 8 + (e - 8)));
  int n    = nt * 16 + (lane & 15);
  float val = (n < Nout) ? W[(size_t)k * Nout + n] : 0.0f;
  frags[idx] = (_Float16)val;
}

// ---------------- fused main kernel ----------------
union AFrag { v16h v; unsigned int u[8]; };
union BFrag { v16h v; uint4 q[2]; };

// Load one 16x32 A fragment (row m) from LDS per ISA 7.12.2 16-bit A layout.
__device__ __forceinline__ v16h load_afrag(const _Float16* inbuf, int inStride,
                                           int m, int hi, int kb) {
  AFrag a;
#pragma unroll
  for (int j = 0; j < 8; ++j) {
    int k0 = kb * 32 + ((j < 4) ? 0 : 16) + hi * 8 + (j & 3) * 2;
    a.u[j] = *reinterpret_cast<const unsigned int*>(&inbuf[m * inStride + k0]);
  }
  return a.v;
}

template<int KB>
__device__ __forceinline__ void mlp_layer(const _Float16* inbuf, int inStride,
                                          const _Float16* frags,
                                          int nt, int lane,
                                          const float* bias,
                                          _Float16* outbuf, int outStride) {
  const int m  = lane & 15;
  const int hi = lane >> 4;
  v8f c = {};
#pragma unroll
  for (int kb = 0; kb < KB; ++kb) {
    v16h a = load_afrag(inbuf, inStride, m, hi, kb);
    BFrag b;
    const uint4* bp = reinterpret_cast<const uint4*>(
        frags + ((size_t)((kb * 8 + nt) * 32 + lane)) * 16);
    b.q[0] = bp[0];
    b.q[1] = bp[1];
    c = __builtin_amdgcn_wmma_f32_16x16x32_f16(false, a, false, b.v,
                                               (short)0, c, false, false);
  }
  const float bn = bias[nt * 16 + m];
#pragma unroll
  for (int r = 0; r < 8; ++r) {
    float x = c[r] + bn;
    float y = x * (1.0f / (1.0f + __expf(-x)));           // SiLU
    outbuf[(r + 8 * hi) * outStride + nt * 16 + m] = (_Float16)y;
  }
}

__global__ __launch_bounds__(256) void fused_kernel(
    const float* __restrict__ node_feats,
    const float* __restrict__ field_feats,
    const float* __restrict__ charges_0,
    const float* __restrict__ charges_ind,
    const float* __restrict__ wsf,
    const unsigned short* __restrict__ fragsRaw,
    const float* __restrict__ b1,
    const float* __restrict__ b2,
    const float* __restrict__ b3,
    const float* __restrict__ b4,
    float* __restrict__ out,
    int N) {
  const _Float16* frags = reinterpret_cast<const _Float16*>(fragsRaw);
  __shared__ _Float16 hA[16 * 264];   // h[16][256], padded stride
  __shared__ _Float16 hB[16 * 136];
  __shared__ _Float16 hC[16 * 136];

  const int t  = threadIdx.x;
  const int nb = blockIdx.x * 16;

  // ---- Stage A: folded tensor-product features -> h[16][256] in LDS (f16)
  // node_feats is a single-use 205 MB stream: use non-temporal loads so the
  // L2-resident weight fragments are not evicted by the stream.
  if (t < 128) {
    const int u = t;
    const float cq  = wsf[u];
    const float cqb = wsf[128 + u];
#pragma unroll 4
    for (int b = 0; b < 16; ++b) {
      int node = nb + b;
      float h = 0.f;
      if (node < N) {
        float s   = __builtin_nontemporal_load(&node_feats[(size_t)node * 512 + u]);
        float qin = __builtin_nontemporal_load(&charges_0[node]) +
                    __builtin_nontemporal_load(&charges_ind[node]);
        h = PATH_W_C * s * (qin * cq + cqb);
      }
      hA[b * 264 + u] = (_Float16)h;
    }
  } else {
    const int u = t - 128;
    const float cv  = wsf[256 + u];
    const float cvb = wsf[384 + u];
    const float cw  = wsf[512 + u] * INV_SQRT3;
#pragma unroll 4
    for (int b = 0; b < 16; ++b) {
      int node = nb + b;
      float h = 0.f;
      if (node < N) {
        const float* nf = node_feats + (size_t)node * 512;
        float s  = __builtin_nontemporal_load(&nf[u]);
        float v0 = __builtin_nontemporal_load(&nf[128 + u * 3 + 0]);
        float v1 = __builtin_nontemporal_load(&nf[128 + u * 3 + 1]);
        float v2 = __builtin_nontemporal_load(&nf[128 + u * 3 + 2]);
        const float* ff = field_feats + (size_t)node * 4;
        float f0 = __builtin_nontemporal_load(&ff[0]);
        float f1 = __builtin_nontemporal_load(&ff[1]);
        float f2 = __builtin_nontemporal_load(&ff[2]);
        float f3 = __builtin_nontemporal_load(&ff[3]);
        h = PATH_W_C * (s * (f0 * cv + cvb) +
                        cw * (v0 * f1 + v1 * f2 + v2 * f3));
      }
      hA[b * 264 + 128 + u] = (_Float16)h;
    }
  }
  __syncthreads();

  const int wave = t >> 5;   // 8 waves, one 16-wide output tile each
  const int lane = t & 31;

  mlp_layer<8>(hA, 264, frags + FR1_OFF, wave, lane, b1, hB, 136);
  __syncthreads();
  mlp_layer<4>(hB, 136, frags + FR2_OFF, wave, lane, b2, hC, 136);
  __syncthreads();
  mlp_layer<4>(hC, 136, frags + FR3_OFF, wave, lane, b3, hB, 136);
  __syncthreads();

  // ---- Layer 4: 128 -> 1 via WMMA on wave 0 using zero-padded W4 fragments.
  // D column 0 (lanes 0 and 16) carries the per-node scalar outputs.
  if (wave == 0) {
    const int m  = lane & 15;
    const int hi = lane >> 4;
    v8f c = {};
#pragma unroll
    for (int kb = 0; kb < 4; ++kb) {
      v16h a = load_afrag(hB, 136, m, hi, kb);
      BFrag b;
      const uint4* bp = reinterpret_cast<const uint4*>(
          frags + FR4_OFF + ((size_t)(kb * 32 + lane)) * 16);
      b.q[0] = bp[0];
      b.q[1] = bp[1];
      c = __builtin_amdgcn_wmma_f32_16x16x32_f16(false, a, false, b.v,
                                                 (short)0, c, false, false);
    }
    if (m == 0) {
      const float bb = b4[0];
#pragma unroll
      for (int r = 0; r < 8; ++r) {
        int node = nb + r + 8 * hi;
        if (node < N) out[node] = c[r] + bb;
      }
    }
  }
}

extern "C" void kernel_launch(void* const* d_in, const int* in_sizes, int n_in,
                              void* d_out, int out_size, void* d_ws, size_t ws_size,
                              hipStream_t stream) {
  (void)n_in; (void)out_size; (void)ws_size;
  const float* node_feats  = (const float*)d_in[1];
  const float* field_feats = (const float*)d_in[5];
  const float* charges_0   = (const float*)d_in[6];
  const float* charges_ind = (const float*)d_in[7];
  const float* Wq0 = (const float*)d_in[8];
  const float* bq  = (const float*)d_in[9];
  const float* Wv0 = (const float*)d_in[10];
  const float* bv  = (const float*)d_in[11];
  const float* Wv1 = (const float*)d_in[12];
  const float* wq_s = (const float*)d_in[13];
  // d_in[14] (wq_v) is dead: it multiplies q_up_v == 0
  const float* wv_s = (const float*)d_in[15];
  const float* wv_v = (const float*)d_in[16];
  const float* W1 = (const float*)d_in[17];
  const float* b1 = (const float*)d_in[18];
  const float* W2 = (const float*)d_in[19];
  const float* b2 = (const float*)d_in[20];
  const float* W3 = (const float*)d_in[21];
  const float* b3 = (const float*)d_in[22];
  const float* W4 = (const float*)d_in[23];
  const float* b4 = (const float*)d_in[24];

  float* wsf = (float*)d_ws;
  unsigned short* frags = (unsigned short*)((char*)d_ws + FRAG_OFF_BYTES);

  int N = in_sizes[1] / 512;   // node_feats is [N, 4*C]

  prep_consts<<<1, 128, 0, stream>>>(wq_s, Wq0, bq, wv_s, Wv0, bv, wv_v, Wv1, wsf);
  prep_frags<<<(FRAG_TOTAL + 255) / 256, 256, 0, stream>>>(W1, W2, W3, W4, frags);
  fused_kernel<<<(N + 15) / 16, 256, 0, stream>>>(
      node_feats, field_feats, charges_0, charges_ind,
      wsf, frags, b1, b2, b3, b4, (float*)d_out, N);
}
